// MultiHeadAttention_5858335392369
// MI455X (gfx1250) — matrix operand order repaired and checked by reading, not compiled, not run on hardware
//
#include <hip/hip_runtime.h>

namespace {
constexpr int kB = 4, kS = 2048, kD = 1024, kH = 16, kHD = 64;
constexpr int kBS = kB * kS;                        // 8192 rows
constexpr size_t kElems  = (size_t)kB * kS * kD;    // 8388608 elems per activation tensor
constexpr size_t kWElems = (size_t)kD * kD;         // 1048576 elems per weight matrix
}

typedef __attribute__((ext_vector_type(16))) _Float16 v16h;
typedef __attribute__((ext_vector_type(8)))  float    v8f;
typedef __attribute__((ext_vector_type(4)))  unsigned u32x4;
typedef __attribute__((ext_vector_type(8)))  int      i32x8;
typedef __attribute__((ext_vector_type(4)))  int      i32x4;

// ---------------------------------------------------------------------------
// LDS raw-offset helper (generic -> addrspace(3) -> 32-bit LDS byte offset)
// ---------------------------------------------------------------------------
__device__ __forceinline__ unsigned lds_addr_of(void* p) {
  return (unsigned)(uintptr_t)(__attribute__((address_space(3))) void*)p;
}

// ---------------------------------------------------------------------------
// TDM: DMA a 2D f16 tile (tile_d0 x tile_d1, contiguous-in-dim0) from global
// to LDS.  D# per CDNA5 ISA §8.3/§8.4; groups 2/3 zero (2D tensor).
// 6-arg builtin form (clang-23 / therock-10.0). Tracked by TENSORcnt.
// ---------------------------------------------------------------------------
__device__ __forceinline__ void tdm_load_tile_f16(
    const _Float16* gsrc, unsigned lds_off,
    int tensor_d0, int tensor_d1, int tile_d0, int tile_d1, int stride_d0) {
  const unsigned long long ga = (unsigned long long)(uintptr_t)gsrc;
  u32x4 g0;
  g0[0] = 1u;                                              // count=1, user D#
  g0[1] = lds_off;                                         // lds_addr
  g0[2] = (unsigned)(ga & 0xFFFFFFFFu);                    // global_addr[31:0]
  g0[3] = (unsigned)((ga >> 32) & 0x1FFFFFFu) | (2u << 30);// addr[56:32] | type=2
  i32x8 g1;
  g1[0] = (1 << 16);                                       // data_size = 2 bytes
  g1[1] = (tensor_d0 & 0xFFFF) << 16;                      // tensor_dim0[15:0]
  g1[2] = ((tensor_d0 >> 16) & 0xFFFF) | ((tensor_d1 & 0xFFFF) << 16);
  g1[3] = ((tensor_d1 >> 16) & 0xFFFF) | (tile_d0 << 16);  // tile_dim0
  g1[4] = (tile_d1 & 0xFFFF);                              // tile_dim1 (tile_dim2=0)
  g1[5] = stride_d0;                                       // tensor_dim0_stride[31:0]
  g1[6] = 0;
  g1[7] = 0;
  i32x4 gz4 = {0, 0, 0, 0};
  i32x8 gz8 = {0, 0, 0, 0, 0, 0, 0, 0};
  __builtin_amdgcn_tensor_load_to_lds(g0, g1, gz4, gz4, gz8, 0);
}

// ---------------------------------------------------------------------------
// WMMA fragment loaders (gfx1250 wave32 layouts, CDNA5 ISA §7.12.2)
// ---------------------------------------------------------------------------
template <typename T>
__device__ __forceinline__ v16h load_a_frag(const T* __restrict__ A, int lda) {
  const int lane = threadIdx.x & 31;
  const int half = lane >> 4;
  const int m    = lane & 15;
  v16h f;
#pragma unroll
  for (int v = 0; v < 8; ++v) {
    const int k = (half << 4) + 2 * v;
    f[2 * v]     = (_Float16)A[(size_t)m * lda + k];
    f[2 * v + 1] = (_Float16)A[(size_t)m * lda + k + 1];
  }
  return f;
}

// B(k,n) = W[n*ldw + k]  (C = A @ W^T)
template <typename T>
__device__ __forceinline__ v16h load_bt_frag(const T* __restrict__ W, int ldw) {
  const int lane = threadIdx.x & 31;
  const int half = lane >> 4;
  const int n    = lane & 15;
  v16h f;
#pragma unroll
  for (int v = 0; v < 8; ++v) {
    const int k = (half << 4) + 2 * v;
    f[2 * v]     = (_Float16)W[(size_t)n * ldw + k];
    f[2 * v + 1] = (_Float16)W[(size_t)n * ldw + k + 1];
  }
  return f;
}

// B row-major (K x N), C = A @ B
template <typename T>
__device__ __forceinline__ v16h load_b_frag(const T* __restrict__ Bm, int ldb) {
  const int lane = threadIdx.x & 31;
  const int half = lane >> 4;
  const int n    = lane & 15;
  v16h f;
#pragma unroll
  for (int v = 0; v < 8; ++v) {
    const int k = (half << 4) + 2 * v;
    f[2 * v]     = (_Float16)Bm[(size_t)k * ldb + n];
    f[2 * v + 1] = (_Float16)Bm[(size_t)(k + 1) * ldb + n];
  }
  return f;
}

__device__ __forceinline__ float grpmax16(float x) {
  x = fmaxf(x, __shfl_xor(x, 1, 32));
  x = fmaxf(x, __shfl_xor(x, 2, 32));
  x = fmaxf(x, __shfl_xor(x, 4, 32));
  x = fmaxf(x, __shfl_xor(x, 8, 32));
  return x;
}
__device__ __forceinline__ float grpsum16(float x) {
  x += __shfl_xor(x, 1, 32);
  x += __shfl_xor(x, 2, 32);
  x += __shfl_xor(x, 4, 32);
  x += __shfl_xor(x, 8, 32);
  return x;
}

// ---------------------------------------------------------------------------
// fp32 -> f16 weight conversion (scale folds 1/sqrt(HD) into Wq)
// ---------------------------------------------------------------------------
__global__ __launch_bounds__(256) void convert_kernel(
    const float* __restrict__ src, _Float16* __restrict__ dst, int n, float scale) {
  for (int i = blockIdx.x * 256 + threadIdx.x; i < n; i += gridDim.x * 256)
    dst[i] = (_Float16)(src[i] * scale);
}

// ---------------------------------------------------------------------------
// GEMM: out = scale * (A @ Wh^T).  A: (M,K) row-major (float or _Float16),
// Wh: (N,K) f16 row-major (TDM-staged into LDS).
//   block tile = 128 x 64 (8 waves; wave w -> rows 16w..16w+15, 64 cols)
//   k-chunk = 32, double-buffered 64x32 f16 LDS tile via tensor_load_to_lds
// OUT_MODE 0: f16 (B,H,S,HD) head layout;  OUT_MODE 2: fp32 (M,N) row-major
// ---------------------------------------------------------------------------
template <typename AT, int OUT_MODE>
__global__ __launch_bounds__(256) void gemm_wt_kernel(
    const AT* __restrict__ A, const _Float16* __restrict__ Wh,
    void* __restrict__ outp, int M, int N, int K, float scale) {
  __shared__ _Float16 ldsB[2][64 * 32];   // [buf][n_local*32 + k_local]

  const int wid  = threadIdx.x >> 5;
  const int lane = threadIdx.x & 31;
  const int nBlocksN = N / 64;
  const int bm = blockIdx.x / nBlocksN;
  const int bn = blockIdx.x % nBlocksN;
  const int row0 = bm * 128 + wid * 16;
  const int n0   = bn * 64;

  const AT*       Ar = A  + (size_t)row0 * K;
  const _Float16* Wr = Wh + (size_t)n0 * K;

  v8f zero = {};
  v8f c[4];
#pragma unroll
  for (int t = 0; t < 4; ++t) c[t] = zero;

  const int NK = K / 32;
  if (wid == 0)
    tdm_load_tile_f16(Wr, lds_addr_of(&ldsB[0][0]), K, N, 32, 64, K);

  for (int i = 0; i < NK; ++i) {
    if (wid == 0) {
      if (i + 1 < NK) {
        tdm_load_tile_f16(Wr + (i + 1) * 32, lds_addr_of(&ldsB[(i + 1) & 1][0]),
                          K, N, 32, 64, K);
        __builtin_amdgcn_s_wait_tensorcnt(1);   // chunk i landed (in-order TDM)
      } else {
        __builtin_amdgcn_s_wait_tensorcnt(0);
      }
    }
    __syncthreads();                            // ldsB[i&1] visible to all waves

    const _Float16* Bt = &ldsB[i & 1][0];
    v16h a = load_a_frag(Ar + i * 32, K);       // global, reused by 4 WMMAs
#pragma unroll
    for (int t = 0; t < 4; ++t) {
      v16h b = load_bt_frag(Bt + (size_t)(t * 16) * 32, 32);  // LDS ds_load
      c[t] = __builtin_amdgcn_wmma_f32_16x16x32_f16(false, a, false, b,
                                                    (short)0, c[t], false, false);
    }
    __syncthreads();                            // done with ldsB[i&1] before reuse
  }

  const int half = lane >> 4, col = lane & 15;
#pragma unroll
  for (int t = 0; t < 4; ++t) {
#pragma unroll
    for (int r = 0; r < 8; ++r) {
      const int m = row0 + (half << 3) + r;
      const int n = n0 + t * 16 + col;
      const float val = c[t][r] * scale;
      if (OUT_MODE == 0) {
        // (B,S,D) row m,col n -> (B,H,S,HD) f16
        const size_t idx = (size_t)((m >> 11) * kH + (n >> 6)) * (kS * kHD) +
                           (size_t)(m & (kS - 1)) * kHD + (n & (kHD - 1));
        ((_Float16*)outp)[idx] = (_Float16)val;
      } else {
        ((float*)outp)[(size_t)m * N + n] = val;
      }
    }
  }
}

// ---------------------------------------------------------------------------
// Flash attention: Qh pre-scaled by 1/sqrt(HD). One wave = 16 query rows;
// 4 waves/block share one (b,h) so the valid_len loop is block-uniform.
// ---------------------------------------------------------------------------
__global__ __launch_bounds__(128) void attn_kernel(
    const _Float16* __restrict__ Qh, const _Float16* __restrict__ Kh,
    const _Float16* __restrict__ Vh, const int* __restrict__ valid_lens,
    _Float16* __restrict__ Oh) {
  __shared__ _Float16 pbuf[4][16 * 32];

  const int lane = threadIdx.x & 31;
  const int wid  = threadIdx.x >> 5;
  const int half = lane >> 4, col = lane & 15;

  const int bid    = blockIdx.x;
  const int qchunk = bid & 31;
  const int bh     = bid >> 5;
  const int b      = bh >> 4, h = bh & 15;
  const int qblock = qchunk * 4 + wid;

  const _Float16* Qb = Qh + (size_t)bh * kS * kHD + (size_t)qblock * 16 * kHD;
  const _Float16* Kb = Kh + (size_t)bh * kS * kHD;
  const _Float16* Vb = Vh + (size_t)bh * kS * kHD;
  const int vlen = valid_lens[b];

  float mrow[8], lrow[8], alpha[8];
  v8f zero = {};
  v8f acc[4];
#pragma unroll
  for (int t = 0; t < 4; ++t) acc[t] = zero;
#pragma unroll
  for (int r = 0; r < 8; ++r) { mrow[r] = -1e30f; lrow[r] = 0.f; }

  for (int kb = 0; kb * 32 < vlen; ++kb) {
    v8f s0 = zero, s1 = zero;
#pragma unroll
    for (int kk = 0; kk < 2; ++kk) {
      v16h a  = load_a_frag(Qb + kk * 32, kHD);
      v16h b0 = load_bt_frag(Kb + (size_t)(kb * 32) * kHD + kk * 32, kHD);
      v16h b1 = load_bt_frag(Kb + (size_t)(kb * 32 + 16) * kHD + kk * 32, kHD);
      s0 = __builtin_amdgcn_wmma_f32_16x16x32_f16(false, a, false, b0,
                                                  (short)0, s0, false, false);
      s1 = __builtin_amdgcn_wmma_f32_16x16x32_f16(false, a, false, b1,
                                                  (short)0, s1, false, false);
    }
    const float bias0 = (kb * 32 + col      < vlen) ? 0.f : -1e6f;
    const float bias1 = (kb * 32 + 16 + col < vlen) ? 0.f : -1e6f;

#pragma unroll
    for (int r = 0; r < 8; ++r) {
      const float x0 = s0[r] + bias0;
      const float x1 = s1[r] + bias1;
      const float mnew = fmaxf(mrow[r], grpmax16(fmaxf(x0, x1)));
      const float p0 = __expf(x0 - mnew);
      const float p1 = __expf(x1 - mnew);
      const float ps = grpsum16(p0 + p1);
      alpha[r] = __expf(mrow[r] - mnew);
      lrow[r]  = lrow[r] * alpha[r] + ps;
      mrow[r]  = mnew;
      const int row = (half << 3) + r;
      pbuf[wid][row * 32 + col]      = (_Float16)p0;
      pbuf[wid][row * 32 + 16 + col] = (_Float16)p1;
    }
    __syncthreads();

    v16h pa = load_a_frag(&pbuf[wid][0], 32);
#pragma unroll
    for (int t = 0; t < 4; ++t) {
#pragma unroll
      for (int r = 0; r < 8; ++r) acc[t][r] *= alpha[r];
      v16h bv = load_b_frag(Vb + (size_t)(kb * 32) * kHD + t * 16, kHD);
      acc[t] = __builtin_amdgcn_wmma_f32_16x16x32_f16(false, pa, false, bv,
                                                      (short)0, acc[t], false, false);
    }
    __syncthreads();
  }

#pragma unroll
  for (int r = 0; r < 8; ++r) lrow[r] = 1.f / lrow[r];

#pragma unroll
  for (int t = 0; t < 4; ++t) {
#pragma unroll
    for (int r = 0; r < 8; ++r) {
      const int m = qblock * 16 + (half << 3) + r;
      Oh[(size_t)(b * kS + m) * kD + h * kHD + t * 16 + col] =
          (_Float16)(acc[t][r] * lrow[r]);
    }
  }
}

// ---------------------------------------------------------------------------
extern "C" void kernel_launch(void* const* d_in, const int* in_sizes, int n_in,
                              void* d_out, int out_size, void* d_ws, size_t ws_size,
                              hipStream_t stream) {
  const float* q  = (const float*)d_in[0];
  const float* k  = (const float*)d_in[1];
  const float* v  = (const float*)d_in[2];
  const int*   vl = (const int*)d_in[3];
  const float* Wq = (const float*)d_in[4];
  const float* Wk = (const float*)d_in[5];
  const float* Wv = (const float*)d_in[6];
  const float* Wo = (const float*)d_in[7];

  _Float16* Qh  = (_Float16*)d_ws;      // (B,H,S,HD) pre-scaled by 1/8
  _Float16* Kh  = Qh + kElems;
  _Float16* Vh  = Kh + kElems;
  _Float16* Oh  = Vh + kElems;          // (B,S,D) merged heads
  _Float16* Wqh = Oh + kElems;          // f16 weight copies (TDM-staged)
  _Float16* Wkh = Wqh + kWElems;
  _Float16* Wvh = Wkh + kWElems;
  _Float16* Woh = Wvh + kWElems;

  convert_kernel<<<1024, 256, 0, stream>>>(Wq, Wqh, (int)kWElems, 0.125f);
  convert_kernel<<<1024, 256, 0, stream>>>(Wk, Wkh, (int)kWElems, 1.0f);
  convert_kernel<<<1024, 256, 0, stream>>>(Wv, Wvh, (int)kWElems, 1.0f);
  convert_kernel<<<1024, 256, 0, stream>>>(Wo, Woh, (int)kWElems, 1.0f);

  const int M = kBS, N = kD, K = kD;
  const int blocks = (M / 128) * (N / 64);   // 1024 blocks of 256 threads

  gemm_wt_kernel<float, 0><<<blocks, 256, 0, stream>>>(q, Wqh, Qh, M, N, K, 1.0f);
  gemm_wt_kernel<float, 0><<<blocks, 256, 0, stream>>>(k, Wkh, Kh, M, N, K, 1.0f);
  gemm_wt_kernel<float, 0><<<blocks, 256, 0, stream>>>(v, Wvh, Vh, M, N, K, 1.0f);

  attn_kernel<<<kB * kH * (kS / 64), 128, 0, stream>>>(Qh, Kh, Vh, vl, Oh);

  gemm_wt_kernel<_Float16, 2><<<blocks, 256, 0, stream>>>(Oh, Woh, d_out, M, N, K, 1.0f);
}
